// MolecularEncoder_39651138076879
// MI455X (gfx1250) — matrix operand order repaired and checked by reading, not compiled, not run on hardware
//
#include <hip/hip_runtime.h>
#include <cstddef>

// ---------------------------------------------------------------------------
// Types for CDNA5 WMMA
// ---------------------------------------------------------------------------
typedef __attribute__((ext_vector_type(16))) _Float16 v16h;
typedef __attribute__((ext_vector_type(8)))  _Float16 v8h;
typedef __attribute__((ext_vector_type(4)))  _Float16 v4h;
typedef __attribute__((ext_vector_type(8)))  float    v8f;

// ---------------------------------------------------------------------------
// Block-cooperative GEMM: C[M,N] = A[M,K] @ B[K,N] (+C if accum)(+bias)(relu)
// f32 in/out, f16 WMMA compute with f32 accumulate.
//
// Block tile 64x64, 256 threads = 8 waves (wave w: wm=w>>2, wn=w&3).
// Wave computes rows [wm*32,+32) x cols [wn*16,+16) as two 16x16 WMMA accs.
// K stepped by 32 through LDS. Interior blocks + full K-steps use float4
// global loads (b128) and packed LDS stores; edges/K-tail use clamped scalar.
//
// Fragment layouts per CDNA5 ISA 7.12.2 (wave32), half = lane>>4:
//  A (16x32 f16): lane row m=lane&15; halfs at k in [8h,8h+8) and [16+8h,+8)
//  B (32x16 f16): lane col n=lane&15; halfs at k in [8h,8h+8) and [16+8h,+8)
//  C/D (16x16 f32): VGPR r: m = r + 8*half, n = lane&15
// ---------------------------------------------------------------------------
__device__ __forceinline__ void wmma_step(const _Float16 (&As)[64][40],
                                          const _Float16 (&Bs)[64][40],
                                          int wm, int wn, int l15, int half,
                                          v8f& acc0, v8f& acc1)
{
    const int cL = wn * 16 + l15;
    v8h b0 = *(const v8h*)&Bs[cL][8 * half];
    v8h b1 = *(const v8h*)&Bs[cL][16 + 8 * half];
    v16h bf = __builtin_shufflevector(b0, b1, 0,1,2,3,4,5,6,7,8,9,10,11,12,13,14,15);
    {
        int rL = wm * 32 + l15;
        v8h a0 = *(const v8h*)&As[rL][8 * half];
        v8h a1 = *(const v8h*)&As[rL][16 + 8 * half];
        v16h af = __builtin_shufflevector(a0, a1, 0,1,2,3,4,5,6,7,8,9,10,11,12,13,14,15);
        acc0 = __builtin_amdgcn_wmma_f32_16x16x32_f16(false, af, false, bf,
                                                      (short)0, acc0, false, false);
    }
    {
        int rL = wm * 32 + 16 + l15;
        v8h a0 = *(const v8h*)&As[rL][8 * half];
        v8h a1 = *(const v8h*)&As[rL][16 + 8 * half];
        v16h af = __builtin_shufflevector(a0, a1, 0,1,2,3,4,5,6,7,8,9,10,11,12,13,14,15);
        acc1 = __builtin_amdgcn_wmma_f32_16x16x32_f16(false, af, false, bf,
                                                      (short)0, acc1, false, false);
    }
}

__global__ __launch_bounds__(256)
void k_wmma_gemm(const float* __restrict__ A, int lda,
                 const float* __restrict__ B, int ldb,
                 float* __restrict__ C, int ldc,
                 int M, int N, int K,
                 const float* __restrict__ bias,
                 int accum, int relu)
{
    __shared__ _Float16 As[64][40];   // 64 rows x 32 k (pad to 40 halfs = 80B)
    __shared__ _Float16 Bs[64][40];   // 64 cols x 32 k (transposed B)

    const int tid  = threadIdx.x;
    const int lane = tid & 31;
    const int w    = tid >> 5;
    const int wm   = w >> 2;          // 0..1
    const int wn   = w & 3;           // 0..3
    const int half = lane >> 4;
    const int l15  = lane & 15;

    const int nt = (N + 63) >> 6;
    const int bm = (int)blockIdx.x / nt;
    const int bn = (int)blockIdx.x % nt;
    const int rowBase = bm * 64;
    const int colBase = bn * 64;
    const int colW = colBase + wn * 16 + l15;   // this lane's C column

    v8f acc0 = {}, acc1 = {};
    if (accum) {
        #pragma unroll
        for (int r = 0; r < 8; ++r) {
            int m0 = rowBase + wm * 32 + r + 8 * half;
            int m1 = m0 + 16;
            bool ok0 = (m0 < M) && (colW < N);
            bool ok1 = (m1 < M) && (colW < N);
            size_t i0 = ok0 ? ((size_t)m0 * ldc + colW) : 0;
            size_t i1 = ok1 ? ((size_t)m1 * ldc + colW) : 0;
            float c0 = C[i0], c1 = C[i1];
            acc0[r] = ok0 ? c0 : 0.f;
            acc1[r] = ok1 ? c1 : 0.f;
        }
    }

    // block-uniform fast-path predicate (full 64x64 tile; lda/ldb mult of 4
    // guaranteed by callers whenever K >= 32)
    const bool fullMN = (rowBase + 64 <= M) && (colBase + 64 <= N);
    const int  kfast  = fullMN ? (K & ~31) : 0;

    // staging coordinates (fixed per thread)
    const int ka = tid & 31, ra0 = tid >> 5;    // A scalar path
    const int cb = tid & 63, rb0 = tid >> 6;    // B scalar path
    const int raf = tid >> 3, qaf = (tid & 7) * 4;     // A fast: row, k-quad
    const int rbf = tid >> 4, qbf = (tid & 15) * 4;    // B fast: k-row, col-quad

    int kk = 0;
    for (; kk < kfast; kk += 32) {              // -------- fast path --------
        __syncthreads();
        #pragma unroll
        for (int p = 0; p < 2; ++p) {
            int r = raf + p * 32;
            const float4 av = *(const float4*)&A[(size_t)(rowBase + r) * lda + kk + qaf];
            v4h hv; hv[0] = (_Float16)av.x; hv[1] = (_Float16)av.y;
                    hv[2] = (_Float16)av.z; hv[3] = (_Float16)av.w;
            *(v4h*)&As[r][qaf] = hv;
        }
        #pragma unroll
        for (int p = 0; p < 2; ++p) {
            int r = rbf + p * 16;
            const float4 bv = *(const float4*)&B[(size_t)(kk + r) * ldb + colBase + qbf];
            Bs[qbf + 0][r] = (_Float16)bv.x;
            Bs[qbf + 1][r] = (_Float16)bv.y;
            Bs[qbf + 2][r] = (_Float16)bv.z;
            Bs[qbf + 3][r] = (_Float16)bv.w;
        }
        __syncthreads();
        wmma_step(As, Bs, wm, wn, l15, half, acc0, acc1);
    }
    for (; kk < K; kk += 32) {                  // ----- clamped tail path ----
        __syncthreads();
        #pragma unroll
        for (int p = 0; p < 8; ++p) {
            int r = ra0 + p * 8;
            int gr = rowBase + r, gk = kk + ka;
            bool ok = (gr < M) && (gk < K);
            size_t idx = ok ? ((size_t)gr * lda + gk) : 0;
            float v = A[idx];
            As[r][ka] = (_Float16)(ok ? v : 0.f);
        }
        #pragma unroll
        for (int p = 0; p < 8; ++p) {
            int r = rb0 + p * 4;
            int gk = kk + r, gc = colBase + cb;
            bool ok = (gk < K) && (gc < N);
            size_t idx = ok ? ((size_t)gk * ldb + gc) : 0;
            float v = B[idx];
            Bs[cb][r] = (_Float16)(ok ? v : 0.f);
        }
        __syncthreads();
        wmma_step(As, Bs, wm, wn, l15, half, acc0, acc1);
    }

    // ---- epilogue
    float bc = 0.f;
    if (bias && colW < N) bc = bias[colW];
    #pragma unroll
    for (int r = 0; r < 8; ++r) {
        int m0 = rowBase + wm * 32 + r + 8 * half;
        int m1 = m0 + 16;
        if (m0 < M && colW < N) {
            float v = acc0[r] + bc;
            if (relu) v = fmaxf(v, 0.f);
            C[(size_t)m0 * ldc + colW] = v;
        }
        if (m1 < M && colW < N) {
            float v = acc1[r] + bc;
            if (relu) v = fmaxf(v, 0.f);
            C[(size_t)m1 * ldc + colW] = v;
        }
    }
}

// ---------------------------------------------------------------------------
// Elementwise / scatter helpers
// ---------------------------------------------------------------------------
__device__ __forceinline__ void atomicMaxF(float* a, float v) {
    if (v >= 0.f) atomicMax((int*)a, __float_as_int(v));
    else          atomicMin((unsigned int*)a, __float_as_uint(v));
}

__global__ void k_fill4(float4* p, int n4, float v) {   // n must be mult of 4
    int t = blockIdx.x * blockDim.x + threadIdx.x;
    if (t < n4) p[t] = make_float4(v, v, v, v);
}

// per-node GAT attention logits: osrc/odst[n*H+h] = sum_c xp[n,h,c]*att[h,c]
__global__ void k_att_logits(const float* __restrict__ xp,
                             const float* __restrict__ as_,
                             const float* __restrict__ ad_,
                             float* __restrict__ osrc, float* __restrict__ odst,
                             int n, int H, int C)
{
    int t = blockIdx.x * blockDim.x + threadIdx.x;
    if (t >= n * H) return;
    int node = t / H, h = t % H;
    const float* xr = xp + (size_t)node * H * C + (size_t)h * C;
    const float* a1 = as_ + (size_t)h * C;
    const float* a2 = ad_ + (size_t)h * C;
    float s1 = 0.f, s2 = 0.f;
    for (int c = 0; c < C; ++c) { float v = xr[c]; s1 += v * a1[c]; s2 += v * a2[c]; }
    osrc[t] = s1; odst[t] = s2;
}

// edge pass 1: leaky-relu logit, store, atomic segment max over dst
__global__ void k_edge_logit_max(const int* __restrict__ src, const int* __restrict__ dst,
                                 int E, int ESL,
                                 const float* __restrict__ asrc, const float* __restrict__ adst,
                                 float* __restrict__ alb, float* __restrict__ mx, int H)
{
    int t = blockIdx.x * blockDim.x + threadIdx.x;
    if (t >= ESL * H) return;
    int e = t / H, h = t % H;
    int s = (e < E) ? src[e] : (e - E);
    int d = (e < E) ? dst[e] : (e - E);
    float al = asrc[s * H + h] + adst[d * H + h];
    al = (al > 0.f) ? al : 0.2f * al;
    alb[t] = al;
    atomicMaxF(&mx[d * H + h], al);
}

// edge pass 2: exp(al - max), store, atomic segment sum
__global__ void k_edge_expsum(const int* __restrict__ dst, int E, int ESL,
                              float* __restrict__ alb, const float* __restrict__ mx,
                              float* __restrict__ ssum, int H)
{
    int t = blockIdx.x * blockDim.x + threadIdx.x;
    if (t >= ESL * H) return;
    int e = t / H, h = t % H;
    int d = (e < E) ? dst[e] : (e - E);
    float ex = expf(alb[t] - mx[d * H + h]);
    alb[t] = ex;
    atomicAdd(&ssum[d * H + h], ex);
}

// edge pass 3: normalize attention weights in place: alb = ex / seg_sum
__global__ void k_edge_norm(const int* __restrict__ dst, int E, int ESL,
                            float* __restrict__ alb, const float* __restrict__ ssum, int H)
{
    int t = blockIdx.x * blockDim.x + threadIdx.x;
    if (t >= ESL * H) return;
    int e = t / H, h = t % H;
    int d = (e < E) ? dst[e] : (e - E);
    alb[t] /= ssum[d * H + h];
}

// edge pass 4: out[dst] += xp[src] * alpha   (float4 per thread; C mult of 4)
__global__ void k_gat_scatter(const int* __restrict__ src, const int* __restrict__ dst,
                              int E, int ESL,
                              const float* __restrict__ xp, const float* __restrict__ alb,
                              float* __restrict__ out, int H, int C)
{
    int F = H * C, Q = F >> 2;
    int t = blockIdx.x * blockDim.x + threadIdx.x;
    if (t >= ESL * Q) return;
    int e = t / Q, f = (t % Q) * 4, h = f / C;
    int s = (e < E) ? src[e] : (e - E);
    int d = (e < E) ? dst[e] : (e - E);
    float alpha = alb[e * H + h];
    const float4 xv = *(const float4*)&xp[(size_t)s * F + f];
    float* o = &out[(size_t)d * F + f];
    atomicAdd(o + 0, xv.x * alpha);
    atomicAdd(o + 1, xv.y * alpha);
    atomicAdd(o + 2, xv.z * alpha);
    atomicAdd(o + 3, xv.w * alpha);
}

// in-degree over dst (incl self loops)
__global__ void k_deg(const int* __restrict__ dst, int E, int ESL, float* __restrict__ deg) {
    int e = blockIdx.x * blockDim.x + threadIdx.x;
    if (e >= ESL) return;
    int d = (e < E) ? dst[e] : (e - E);
    atomicAdd(&deg[d], 1.f);
}

// deg -> rsqrt(max(deg,1)) in place
__global__ void k_dis(float* __restrict__ deg, int N) {
    int n = blockIdx.x * blockDim.x + threadIdx.x;
    if (n < N) deg[n] = rsqrtf(fmaxf(deg[n], 1.f));
}

// GCN scatter: out[d] += hg[s] * dis[s] * dis[d]   (float4 per thread)
__global__ void k_gcn_scatter(const int* __restrict__ src, const int* __restrict__ dst,
                              int E, int ESL,
                              const float* __restrict__ hg, const float* __restrict__ dis,
                              float* __restrict__ out, int C)
{
    int Q = C >> 2;
    int t = blockIdx.x * blockDim.x + threadIdx.x;
    if (t >= ESL * Q) return;
    int e = t / Q, c = (t % Q) * 4;
    int s = (e < E) ? src[e] : (e - E);
    int d = (e < E) ? dst[e] : (e - E);
    float norm = dis[s] * dis[d];
    const float4 xv = *(const float4*)&hg[(size_t)s * C + c];
    float* o = &out[(size_t)d * C + c];
    atomicAdd(o + 0, xv.x * norm);
    atomicAdd(o + 1, xv.y * norm);
    atomicAdd(o + 2, xv.z * norm);
    atomicAdd(o + 3, xv.w * norm);
}

__global__ void k_cnt(const int* __restrict__ dst, int E, float* __restrict__ cnt) {
    int e = blockIdx.x * blockDim.x + threadIdx.x;
    if (e < E) atomicAdd(&cnt[dst[e]], 1.f);
}

__global__ void k_sage_sum(const int* __restrict__ src, const int* __restrict__ dst, int E,
                           const float* __restrict__ h, float* __restrict__ sum, int C)
{
    int Q = C >> 2;
    int t = blockIdx.x * blockDim.x + threadIdx.x;
    if (t >= E * Q) return;
    int e = t / Q, c = (t % Q) * 4;
    int s = src[e], d = dst[e];
    const float4 xv = *(const float4*)&h[(size_t)s * C + c];
    float* o = &sum[(size_t)d * C + c];
    atomicAdd(o + 0, xv.x);
    atomicAdd(o + 1, xv.y);
    atomicAdd(o + 2, xv.z);
    atomicAdd(o + 3, xv.w);
}

__global__ void k_sage_mean(float* __restrict__ sum, const float* __restrict__ cnt, int N, int C) {
    int t = blockIdx.x * blockDim.x + threadIdx.x;
    if (t >= N * C) return;
    sum[t] /= fmaxf(cnt[t / C], 1.f);
}

// x = elu( g*((x + addb) - m)*rsqrt(v+eps) + b ), in place; addb may be null
__global__ void k_bn_elu(float* __restrict__ x, const float* __restrict__ addb,
                         const float* __restrict__ g, const float* __restrict__ b,
                         const float* __restrict__ m, const float* __restrict__ v,
                         int n, int F)
{
    int t = blockIdx.x * blockDim.x + threadIdx.x;
    if (t >= n) return;
    int c = t % F;
    float val = x[t] + (addb ? addb[c] : 0.f);
    val = g[c] * (val - m[c]) * rsqrtf(v[c] + 1e-5f) + b[c];
    x[t] = (val > 0.f) ? val : expm1f(val);
}

// GAT2 finalize: mean over heads + bias, then BN+ELU -> node output
__global__ void k_headmean_bn_elu(const float* __restrict__ out4, const float* __restrict__ bias,
                                  const float* __restrict__ g, const float* __restrict__ b,
                                  const float* __restrict__ m, const float* __restrict__ v,
                                  float* __restrict__ node, int N, int H, int C)
{
    int t = blockIdx.x * blockDim.x + threadIdx.x;
    if (t >= N * C) return;
    int n = t / C, c = t % C;
    float s = 0.f;
    for (int h = 0; h < H; ++h) s += out4[(size_t)n * H * C + (size_t)h * C + c];
    s = s / (float)H + bias[c];
    s = g[c] * (s - m[c]) * rsqrtf(v[c] + 1e-5f) + b[c];
    node[t] = (s > 0.f) ? s : expm1f(s);
}

__global__ void k_pool_cnt(const int* __restrict__ batch, int N, float* __restrict__ gcnt) {
    int n = blockIdx.x * blockDim.x + threadIdx.x;
    if (n < N) atomicAdd(&gcnt[batch[n]], 1.f);
}

__global__ void k_pool_sum(const int* __restrict__ batch, const float* __restrict__ node,
                           float* __restrict__ gsum, int N, int C)
{
    int Q = C >> 2;
    int t = blockIdx.x * blockDim.x + threadIdx.x;
    if (t >= N * Q) return;
    int n = t / Q, c = (t % Q) * 4;
    int gidx = batch[n];
    const float4 xv = *(const float4*)&node[(size_t)n * C + c];
    float* o = &gsum[(size_t)gidx * C + c];
    atomicAdd(o + 0, xv.x);
    atomicAdd(o + 1, xv.y);
    atomicAdd(o + 2, xv.z);
    atomicAdd(o + 3, xv.w);
}

__global__ void k_pool_div(const float* __restrict__ gsum, const float* __restrict__ gcnt,
                           float* __restrict__ graph, int G, int C)
{
    int t = blockIdx.x * blockDim.x + threadIdx.x;
    if (t >= G * C) return;
    graph[t] = gsum[t] / fmaxf(gcnt[t / C], 1.f);
}

// ---------------------------------------------------------------------------
// Host orchestration
// ---------------------------------------------------------------------------
extern "C" void kernel_launch(void* const* d_in, const int* in_sizes, int n_in,
                              void* d_out, int out_size, void* d_ws, size_t ws_size,
                              hipStream_t stream)
{
    constexpr int N = 25000, E = 200000, G = 512, FIN = 15, C = 128, H1 = 8, H2 = 4;
    constexpr int F1 = H1 * C;          // 1024
    constexpr int F2 = H2 * C;          // 512
    constexpr int ESL = E + N;          // edges + self loops

    const float* x     = (const float*)d_in[0];
    const int*   ei    = (const int*)  d_in[1];
    const int*   src   = ei;
    const int*   dst   = ei + E;
    const int*   batch = (const int*)  d_in[2];
    const float* W1    = (const float*)d_in[3];
    const float* as1   = (const float*)d_in[4];
    const float* ad1   = (const float*)d_in[5];
    const float* bias1 = (const float*)d_in[6];
    const float* g1 = (const float*)d_in[7],  *b1 = (const float*)d_in[8];
    const float* m1 = (const float*)d_in[9],  *v1 = (const float*)d_in[10];
    const float* Wg = (const float*)d_in[11], *bg = (const float*)d_in[12];
    const float* g2 = (const float*)d_in[13], *b2 = (const float*)d_in[14];
    const float* m2 = (const float*)d_in[15], *v2 = (const float*)d_in[16];
    const float* Wl = (const float*)d_in[17], *bl = (const float*)d_in[18];
    const float* Wr = (const float*)d_in[19];
    const float* g3 = (const float*)d_in[20], *b3 = (const float*)d_in[21];
    const float* m3 = (const float*)d_in[22], *v3 = (const float*)d_in[23];
    const float* W2 = (const float*)d_in[24];
    const float* as2 = (const float*)d_in[25], *ad2 = (const float*)d_in[26];
    const float* bias2 = (const float*)d_in[27];
    const float* g4 = (const float*)d_in[28], *b4 = (const float*)d_in[29];
    const float* m4 = (const float*)d_in[30], *v4 = (const float*)d_in[31];
    const float* Wp1 = (const float*)d_in[32], *bp1 = (const float*)d_in[33];
    const float* Wp2 = (const float*)d_in[34], *bp2 = (const float*)d_in[35];

    // ---- workspace carving (floats; every buffer stays 16B aligned) ----
    float* W = (float*)d_ws;
    size_t p = 0;
    auto alloc = [&](size_t n) { float* r = W + p; p += n; return r; };
    float* regA = alloc((size_t)N * F1);   // GAT1 xp, reused by later layers
    float* regB = alloc((size_t)N * F1);   // GAT1 out / h1, reused by GAT2 out
    float* asrc1 = alloc((size_t)N * H1);
    float* adst1 = alloc((size_t)N * H1);
    float* mx1   = alloc((size_t)N * H1);
    float* sum1  = alloc((size_t)N * H1);
    float* alb1  = alloc((size_t)ESL * H1);
    float* deg   = alloc(N);
    float* cnt   = alloc(N);
    float* asrc2 = alloc((size_t)N * H2);
    float* adst2 = alloc((size_t)N * H2);
    float* mx2   = alloc((size_t)N * H2);
    float* sum2  = alloc((size_t)N * H2);
    float* alb2  = alloc((size_t)ESL * H2);
    float* gsum  = alloc((size_t)G * C);
    float* gcnt  = alloc(G);
    float* ptmp  = alloc((size_t)G * C);
    (void)ws_size; (void)in_sizes; (void)n_in; (void)out_size;

    // aliases into regA/regB across the pipeline
    float* xp1  = regA;                      // [N,1024]  GAT1 linear
    float* out1 = regB;                      // [N,1024]  GAT1 agg -> h1
    float* hg   = regA;                      // [N,128]   GCN linear (xp1 dead)
    float* out2 = regA + (size_t)N * C;      // [N,128]   GCN agg -> h2
    float* sg   = regA + (size_t)2 * N * C;  // [N,128]   SAGE mean
    float* out3 = regA + (size_t)3 * N * C;  // [N,128]   SAGE out -> h3
    float* xp2  = regA + (size_t)4 * N * C;  // [N,512]   GAT2 linear
    float* out4 = regB;                      // [N,512]   GAT2 agg (h1 dead)

    float* nodeO  = (float*)d_out;                 // [N,128]
    float* graphO = nodeO + (size_t)N * C;         // [G,128]
    float* projO  = graphO + (size_t)G * C;        // [G,64]

    const int TB = 256;
    auto nb = [](long n) { return (unsigned)((n + 255) / 256); };
    auto fill = [&](float* ptr, long n, float v) {  // n multiple of 4 (all are)
        k_fill4<<<nb(n / 4), TB, 0, stream>>>((float4*)ptr, (int)(n / 4), v);
    };
    auto gemm = [&](const float* A, int lda, const float* B_, int ldb, float* Cc, int ldc,
                    int M, int Nn, int K, const float* bias, int accum, int relu) {
        long blocks = (long)((M + 63) / 64) * ((Nn + 63) / 64);
        k_wmma_gemm<<<dim3((unsigned)blocks), TB, 0, stream>>>(
            A, lda, B_, ldb, Cc, ldc, M, Nn, K, bias, accum, relu);
    };

    // ---------------- GAT1 ----------------
    gemm(x, FIN, W1, F1, xp1, F1, N, F1, FIN, nullptr, 0, 0);
    k_att_logits<<<nb((long)N * H1), TB, 0, stream>>>(xp1, as1, ad1, asrc1, adst1, N, H1, C);
    fill(mx1, (long)N * H1, -3.0e38f);
    fill(sum1, (long)N * H1, 0.f);
    k_edge_logit_max<<<nb((long)ESL * H1), TB, 0, stream>>>(src, dst, E, ESL, asrc1, adst1, alb1, mx1, H1);
    k_edge_expsum<<<nb((long)ESL * H1), TB, 0, stream>>>(dst, E, ESL, alb1, mx1, sum1, H1);
    k_edge_norm<<<nb((long)ESL * H1), TB, 0, stream>>>(dst, E, ESL, alb1, sum1, H1);
    fill(out1, (long)N * F1, 0.f);
    k_gat_scatter<<<nb((long)ESL * F1 / 4), TB, 0, stream>>>(src, dst, E, ESL, xp1, alb1, out1, H1, C);
    k_bn_elu<<<nb((long)N * F1), TB, 0, stream>>>(out1, bias1, g1, b1, m1, v1, N * F1, F1);

    // ---------------- GCN ----------------
    fill(deg, N, 0.f);
    k_deg<<<nb(ESL), TB, 0, stream>>>(dst, E, ESL, deg);
    k_dis<<<nb(N), TB, 0, stream>>>(deg, N);
    gemm(out1, F1, Wg, C, hg, C, N, C, F1, nullptr, 0, 0);         // xp1 region reused
    fill(out2, (long)N * C, 0.f);
    k_gcn_scatter<<<nb((long)ESL * C / 4), TB, 0, stream>>>(src, dst, E, ESL, hg, deg, out2, C);
    k_bn_elu<<<nb((long)N * C), TB, 0, stream>>>(out2, bg, g2, b2, m2, v2, N * C, C);

    // ---------------- SAGE ----------------
    fill(cnt, N, 0.f);
    k_cnt<<<nb(E), TB, 0, stream>>>(dst, E, cnt);
    fill(sg, (long)N * C, 0.f);
    k_sage_sum<<<nb((long)E * C / 4), TB, 0, stream>>>(src, dst, E, out2, sg, C);
    k_sage_mean<<<nb((long)N * C), TB, 0, stream>>>(sg, cnt, N, C);
    gemm(sg, C, Wl, C, out3, C, N, C, C, bl, 0, 0);                // mean @ Wl + bl
    gemm(out2, C, Wr, C, out3, C, N, C, C, nullptr, 1, 0);         // += h2 @ Wr
    k_bn_elu<<<nb((long)N * C), TB, 0, stream>>>(out3, nullptr, g3, b3, m3, v3, N * C, C);

    // ---------------- GAT2 ----------------
    gemm(out3, C, W2, F2, xp2, F2, N, F2, C, nullptr, 0, 0);
    k_att_logits<<<nb((long)N * H2), TB, 0, stream>>>(xp2, as2, ad2, asrc2, adst2, N, H2, C);
    fill(mx2, (long)N * H2, -3.0e38f);
    fill(sum2, (long)N * H2, 0.f);
    k_edge_logit_max<<<nb((long)ESL * H2), TB, 0, stream>>>(src, dst, E, ESL, asrc2, adst2, alb2, mx2, H2);
    k_edge_expsum<<<nb((long)ESL * H2), TB, 0, stream>>>(dst, E, ESL, alb2, mx2, sum2, H2);
    k_edge_norm<<<nb((long)ESL * H2), TB, 0, stream>>>(dst, E, ESL, alb2, sum2, H2);
    fill(out4, (long)N * F2, 0.f);
    k_gat_scatter<<<nb((long)ESL * F2 / 4), TB, 0, stream>>>(src, dst, E, ESL, xp2, alb2, out4, H2, C);
    k_headmean_bn_elu<<<nb((long)N * C), TB, 0, stream>>>(out4, bias2, g4, b4, m4, v4, nodeO, N, H2, C);

    // ---------------- pooling + projection ----------------
    fill(gcnt, G, 0.f);
    fill(gsum, (long)G * C, 0.f);
    k_pool_cnt<<<nb(N), TB, 0, stream>>>(batch, N, gcnt);
    k_pool_sum<<<nb((long)N * C / 4), TB, 0, stream>>>(batch, nodeO, gsum, N, C);
    k_pool_div<<<nb((long)G * C), TB, 0, stream>>>(gsum, gcnt, graphO, G, C);
    gemm(graphO, C, Wp1, C, ptmp, C, G, C, C, bp1, 0, 1);          // relu(graph@Wp1+bp1)
    gemm(ptmp, C, Wp2, 64, projO, 64, G, 64, C, bp2, 0, 0);        // @Wp2 + bp2
}
